// MoEBlock_7267084665020
// MI455X (gfx1250) — matrix operand order repaired and checked by reading, not compiled, not run on hardware
//
#include <hip/hip_runtime.h>
#include <hip/hip_bf16.h>
#include <math.h>

// ---- problem constants (match reference) ----
#define SQ 2048   // sequence length S
#define DM 1024   // model dim D
#define NH 16     // heads
#define HD 64     // head dim
#define NE 8      // experts
#define FF 2048   // ffn dim F
#define Z_COEF 0.001f
#define LB_COEF 0.01f

typedef __bf16 bf16;
typedef __attribute__((ext_vector_type(16))) __bf16 v16bf;
typedef __attribute__((ext_vector_type(8)))  float  v8f;

__device__ __forceinline__ bf16  f2bf(float x) { return (bf16)x; }
__device__ __forceinline__ float bf2f(bf16 x)  { return (float)x; }

// ---------------------------------------------------------------------------
// Generic WMMA GEMM: C[M,N](f32) = alpha * A[M,K](bf16) @ B[K,N](bf16)
// TRANS_B: B element (k,n) = B[n*ldb + k]  (used for Q@K^T)
// MREP: 16-row tiles per wave (register blocking in M).
// Block = 256 threads (8 waves). Block tile (8*16*MREP) x 64, K step 32.
// B tile double-buffered in LDS with bank-conflict padding. B fragments are
// software-pipelined 2 deep: ds_load of fragment t+1 issues before the WMMA
// burst of fragment t, so LDS latency overlaps the matrix pipe.
// Epilogue supports accumulate and a per-row scale (MoE combine weights).
// ---------------------------------------------------------------------------
#define BN 64
#define BK 32
#define BKP 40   // padded column stride (elements): 80B keeps 16B alignment,
                 // maps the 16 columns of a half-wave to distinct banks

template <int MREP, bool TRANS_B>
__global__ __launch_bounds__(256) void k_gemm(
    const bf16* __restrict__ A, int lda,
    const bf16* __restrict__ B, int ldb,
    float* __restrict__ C, int ldc,
    int M, int N, int K, float alpha,
    const float* __restrict__ rowScale, int rsStride, int accumulate)
{
    constexpr int BM = 8 * 16 * MREP;
    __shared__ bf16 Bs[2][BN * BKP];

    const int tid  = threadIdx.x;
    const int wave = tid >> 5;
    const int lane = tid & 31;
    const int laneHalf = lane >> 4;     // 0 or 1
    const int lm = lane & 15;

    const int m0 = blockIdx.y * BM + wave * (16 * MREP);
    const int n0 = blockIdx.x * BN;

    v8f acc[MREP][4] = {};

    // stage one BK x BN tile of B into LDS (column-major, padded)
    auto stageB = [&](int kt, bf16* dst) {
        if (!TRANS_B) {
            const int k  = tid >> 3;          // 0..31
            const int nb = (tid & 7) * 8;     // 0..56
            const bf16* src = B + (size_t)(kt + k) * ldb + (n0 + nb);
            if (kt + BK < K) __builtin_prefetch(src + (size_t)BK * ldb, 0, 1);
#pragma unroll
            for (int j = 0; j < 8; ++j) dst[(nb + j) * BKP + k] = src[j];
        } else {
            const int n  = tid >> 2;          // 0..63
            const int kk = (tid & 3) * 8;     // 0,8,16,24
            const bf16* src = B + (size_t)(n0 + n) * ldb + (kt + kk);
            if (kt + BK < K) __builtin_prefetch(src + BK, 0, 1);
#pragma unroll
            for (int j = 0; j < 8; ++j) dst[n * BKP + kk + j] = src[j];
        }
    };

    // load one 32x16 B fragment (col = lane%16, K = laneHalf*8 + {0..7, 16..23})
    auto loadBfrag = [&](const bf16* buf, int t) {
        v16bf v;
        const bf16* bcol = &buf[(t * 16 + lm) * BKP + laneHalf * 8];
#pragma unroll
        for (int j = 0; j < 8; ++j) { v[j] = bcol[j]; v[8 + j] = bcol[16 + j]; }
        return v;
    };

    stageB(0, Bs[0]);
    __syncthreads();

    int cur = 0;
    for (int kt = 0; kt < K; kt += BK) {
        // prefetch next B tile into the other buffer while computing this one
        if (kt + BK < K) stageB(kt + BK, Bs[cur ^ 1]);

        // ---- A fragments (16x32 bf16, ISA layout) ----
        // lanes 0-15: K = {0..7, 16..23}; lanes 16-31: K = {8..15, 24..31}
        v16bf af[MREP];
#pragma unroll
        for (int r = 0; r < MREP; ++r) {
            const bf16* arow =
                A + (size_t)(m0 + r * 16 + lm) * lda + kt + laneHalf * 8;
#pragma unroll
            for (int j = 0; j < 8; ++j) { af[r][j] = arow[j]; af[r][8 + j] = arow[16 + j]; }
        }

        // ---- B fragments, 2-deep software pipeline over the 4 N-tiles ----
        v16bf bcur = loadBfrag(Bs[cur], 0);
#pragma unroll
        for (int t = 0; t < 4; ++t) {
            v16bf bnext;
            if (t < 3) bnext = loadBfrag(Bs[cur], t + 1);   // issue before WMMA burst
#pragma unroll
            for (int r = 0; r < MREP; ++r)
                acc[r][t] = __builtin_amdgcn_wmma_f32_16x16x32_bf16(
                    false, af[r], false, bcur, (short)0, acc[r][t], false, false);
            if (t < 3) bcur = bnext;
        }

        __syncthreads();
        cur ^= 1;
    }

    // ---- epilogue: C/D layout: col = lane%16, row(VGPR j) = j + 8*laneHalf ----
#pragma unroll
    for (int r = 0; r < MREP; ++r)
#pragma unroll
        for (int t = 0; t < 4; ++t) {
            const int n = n0 + t * 16 + lm;
#pragma unroll
            for (int j = 0; j < 8; ++j) {
                const int m = m0 + r * 16 + laneHalf * 8 + j;
                float v = alpha * acc[r][t][j];
                if (rowScale) v *= rowScale[(size_t)m * rsStride];
                const size_t idx = (size_t)m * ldc + n;
                if (accumulate) C[idx] += v; else C[idx] = v;
            }
        }
}

// ---------------------------------------------------------------------------
// Elementwise / reduction helper kernels
// ---------------------------------------------------------------------------
__global__ void k_cast_bf16(const float* __restrict__ src, bf16* __restrict__ dst, size_t n) {
    for (size_t i = (size_t)blockIdx.x * blockDim.x + threadIdx.x; i < n;
         i += (size_t)gridDim.x * blockDim.x)
        dst[i] = f2bf(src[i]);
}

__global__ void k_zero_f32(float* __restrict__ p, size_t n) {
    for (size_t i = (size_t)blockIdx.x * blockDim.x + threadIdx.x; i < n;
         i += (size_t)gridDim.x * blockDim.x)
        p[i] = 0.0f;
}

__global__ void k_copy_f32(const float* __restrict__ s, float* __restrict__ d, size_t n) {
    for (size_t i = (size_t)blockIdx.x * blockDim.x + threadIdx.x; i < n;
         i += (size_t)gridDim.x * blockDim.x)
        d[i] = s[i];
}

// LayerNorm row kernel: one block per row, output bf16
__global__ __launch_bounds__(256) void k_layernorm(
    const float* __restrict__ x, const float* __restrict__ w,
    const float* __restrict__ b, bf16* __restrict__ out)
{
    const int m = blockIdx.x, t = threadIdx.x;
    const float* row = x + (size_t)m * DM;
    float s = 0.f, s2 = 0.f;
    for (int d = t; d < DM; d += 256) { float v = row[d]; s += v; s2 += v * v; }
    __shared__ float rs[256], rs2[256];
    rs[t] = s; rs2[t] = s2; __syncthreads();
    for (int off = 128; off > 0; off >>= 1) {
        if (t < off) { rs[t] += rs[t + off]; rs2[t] += rs2[t + off]; }
        __syncthreads();
    }
    const float mean = rs[0] / DM;
    const float var  = rs2[0] / DM - mean * mean;
    const float rstd = rsqrtf(var + 1e-5f);
    for (int d = t; d < DM; d += 256)
        out[(size_t)m * DM + d] = f2bf((row[d] - mean) * rstd * w[d] + b[d]);
}

// RoPE on Q,K (f32 in, bf16 out). One block per sequence position.
__global__ __launch_bounds__(256) void k_rope(
    const float* __restrict__ q, const float* __restrict__ k,
    const int* __restrict__ pos, bf16* __restrict__ qb, bf16* __restrict__ kb)
{
    const int m = blockIdx.x;
    const float p = (float)pos[m];
    for (int d = threadIdx.x; d < DM; d += 256) {
        const int di   = d & (HD - 1);          // index within head
        const int d2   = di & (HD / 2 - 1);     // rotary pair index
        const int base = d & ~(HD - 1);         // head base column
        const float inv = __powf(10000.0f, -(float)d2 / (HD / 2));
        const float fr = p * inv, c = __cosf(fr), s = __sinf(fr);
        const size_t idx = (size_t)m * DM + d;
        const float qv = q[idx], kv = k[idx];
        const size_t ridx = (size_t)m * DM + base + ((di < HD / 2) ? di + HD / 2 : di - HD / 2);
        const float qr = (di < HD / 2) ? -q[ridx] : q[ridx];
        const float kr = (di < HD / 2) ? -k[ridx] : k[ridx];
        qb[idx] = f2bf(qv * c + qr * s);
        kb[idx] = f2bf(kv * c + kr * s);
    }
}

// Causal softmax over one score row (row i valid for columns 0..i), bf16 out.
__global__ __launch_bounds__(256) void k_softmax_causal(
    const float* __restrict__ scores, bf16* __restrict__ probs)
{
    const int i = blockIdx.x, t = threadIdx.x;
    const int nvalid = i + 1;
    const float* row = scores + (size_t)i * SQ;
    __shared__ float red[256];
    float mx = -1e30f;
    for (int j = t; j < nvalid; j += 256) mx = fmaxf(mx, row[j]);
    red[t] = mx; __syncthreads();
    for (int off = 128; off > 0; off >>= 1) {
        if (t < off) red[t] = fmaxf(red[t], red[t + off]);
        __syncthreads();
    }
    mx = red[0]; __syncthreads();
    float sum = 0.f;
    for (int j = t; j < nvalid; j += 256) sum += __expf(row[j] - mx);
    red[t] = sum; __syncthreads();
    for (int off = 128; off > 0; off >>= 1) {
        if (t < off) red[t] += red[t + off];
        __syncthreads();
    }
    const float inv = 1.0f / red[0];
    bf16* prow = probs + (size_t)i * SQ;
    for (int j = t; j < SQ; j += 256)
        prow[j] = f2bf((j < nvalid) ? __expf(row[j] - mx) * inv : 0.0f);
}

// Router: logits = x@wr, softmax over 8, top-2 combine weights + aux partials.
// red layout: red[0]=sum lse^2, red[1..8]=util counts, red[9..16]=prob sums
__global__ __launch_bounds__(256) void k_router(
    const bf16* __restrict__ x, const float* __restrict__ wr,
    float* __restrict__ combine, float* __restrict__ red)
{
    const int m = blockIdx.x, t = threadIdx.x;
    const int e = t >> 5, lane = t & 31;
    float p = 0.f;
    for (int d = lane; d < DM; d += 32)
        p += bf2f(x[(size_t)m * DM + d]) * wr[(size_t)d * NE + e];
    __shared__ float r[256];
    __shared__ float logits[NE];
    r[t] = p; __syncthreads();
    for (int off = 16; off > 0; off >>= 1) {
        if (lane < off) r[t] += r[t + off];
        __syncthreads();
    }
    if (lane == 0) logits[e] = r[t];
    __syncthreads();
    if (t == 0) {
        float mx = logits[0];
        for (int i = 1; i < NE; ++i) mx = fmaxf(mx, logits[i]);
        float pr[NE], se = 0.f;
        for (int i = 0; i < NE; ++i) { pr[i] = __expf(logits[i] - mx); se += pr[i]; }
        const float lse = mx + __logf(se);
        for (int i = 0; i < NE; ++i) pr[i] /= se;
        int i1 = 0;
        for (int i = 1; i < NE; ++i) if (pr[i] > pr[i1]) i1 = i;
        int i2 = (i1 == 0) ? 1 : 0;
        for (int i = 0; i < NE; ++i) if (i != i1 && pr[i] > pr[i2]) i2 = i;
        const float wsum = pr[i1] + pr[i2];
        float* crow = combine + (size_t)m * NE;
        for (int i = 0; i < NE; ++i) crow[i] = 0.f;
        crow[i1] = pr[i1] / wsum;
        crow[i2] = pr[i2] / wsum;
        atomicAdd(&red[0], lse * lse);
        atomicAdd(&red[1 + i1], 1.0f);
        atomicAdd(&red[1 + i2], 1.0f);
        for (int i = 0; i < NE; ++i) atomicAdd(&red[9 + i], pr[i]);
    }
}

__global__ void k_silu_mul(const float* __restrict__ g, const float* __restrict__ u,
                           bf16* __restrict__ h, size_t n) {
    for (size_t i = (size_t)blockIdx.x * blockDim.x + threadIdx.x; i < n;
         i += (size_t)gridDim.x * blockDim.x) {
        const float gv = g[i];
        h[i] = f2bf(gv / (1.0f + __expf(-gv)) * u[i]);
    }
}

__global__ void k_finalize(const float* __restrict__ hs1, const float* __restrict__ moe,
                           float* __restrict__ out, size_t n) {
    for (size_t i = (size_t)blockIdx.x * blockDim.x + threadIdx.x; i < n;
         i += (size_t)gridDim.x * blockDim.x)
        out[i] = hs1[i] + moe[i];
}

__global__ void k_aux(const float* __restrict__ red, float* __restrict__ out) {
    const float T = (float)SQ;
    float z = Z_COEF * (red[0] / T);
    float lb = 0.f;
    for (int e = 0; e < NE; ++e) lb += (red[1 + e] / T) * (red[9 + e] / T);
    out[(size_t)SQ * DM] = z + LB_COEF * lb;
}

// ---------------------------------------------------------------------------
// host launcher
// ---------------------------------------------------------------------------
extern "C" void kernel_launch(void* const* d_in, const int* in_sizes, int n_in,
                              void* d_out, int out_size, void* d_ws, size_t ws_size,
                              hipStream_t stream)
{
    (void)in_sizes; (void)n_in; (void)out_size; (void)ws_size;

    const float* hs    = (const float*)d_in[0];
    // d_in[1] = attention_mask (bool) -- causality computed from indices instead
    const int*   pos   = (const int*)d_in[2];
    const float* ln1w  = (const float*)d_in[3];
    const float* ln1b  = (const float*)d_in[4];
    const float* ln2w  = (const float*)d_in[5];
    const float* ln2b  = (const float*)d_in[6];
    const float* wq    = (const float*)d_in[7];
    const float* wk    = (const float*)d_in[8];
    const float* wv    = (const float*)d_in[9];
    const float* wo    = (const float*)d_in[10];
    const float* wr    = (const float*)d_in[11];
    const float* wg    = (const float*)d_in[12];
    const float* wu    = (const float*)d_in[13];
    const float* wd    = (const float*)d_in[14];
    float* out = (float*)d_out;

    // deterministic bump allocator over d_ws
    char* base = (char*)d_ws;
    size_t off = 0;
    auto alloc = [&](size_t bytes) -> void* {
        void* p = base + off;
        off = (off + bytes + 255) & ~(size_t)255;
        return p;
    };

    const size_t DD  = (size_t)DM * DM;       // 1M
    const size_t SD  = (size_t)SQ * DM;       // 2M
    const size_t SS  = (size_t)SQ * SQ;       // 4M
    const size_t SF  = (size_t)SQ * FF;       // 4M
    const size_t EDF = (size_t)NE * DM * FF;  // 16M

    bf16* wqb = (bf16*)alloc(DD * 2);
    bf16* wkb = (bf16*)alloc(DD * 2);
    bf16* wvb = (bf16*)alloc(DD * 2);
    bf16* wob = (bf16*)alloc(DD * 2);
    bf16* wgb = (bf16*)alloc(EDF * 2);
    bf16* wub = (bf16*)alloc(EDF * 2);
    bf16* wdb = (bf16*)alloc(EDF * 2);

    bf16*  x1b    = (bf16*)alloc(SD * 2);
    float* qf     = (float*)alloc(SD * 4);
    float* kf     = (float*)alloc(SD * 4);
    float* vf     = (float*)alloc(SD * 4);
    bf16*  qb     = (bf16*)alloc(SD * 2);
    bf16*  kb     = (bf16*)alloc(SD * 2);
    bf16*  vb     = (bf16*)alloc(SD * 2);
    float* scores = (float*)alloc(SS * 4);
    bf16*  probs  = (bf16*)alloc(SS * 2);
    float* attnO  = (float*)alloc(SD * 4);
    bf16*  attnOb = (bf16*)alloc(SD * 2);
    float* hs1    = (float*)alloc(SD * 4);
    bf16*  x2b    = (bf16*)alloc(SD * 2);
    float* combine= (float*)alloc((size_t)SQ * NE * 4);
    float* red    = (float*)alloc(32 * 4);
    float* gbuf   = (float*)alloc(SF * 4);
    float* ubuf   = (float*)alloc(SF * 4);
    bf16*  hb     = (bf16*)alloc(SF * 2);
    float* moe    = (float*)alloc(SD * 4);

    const dim3 blk(256);
    const dim3 gcast(2048);

    // ---- weight casts to bf16 ----
    k_cast_bf16<<<gcast, blk, 0, stream>>>(wq, wqb, DD);
    k_cast_bf16<<<gcast, blk, 0, stream>>>(wk, wkb, DD);
    k_cast_bf16<<<gcast, blk, 0, stream>>>(wv, wvb, DD);
    k_cast_bf16<<<gcast, blk, 0, stream>>>(wo, wob, DD);
    k_cast_bf16<<<gcast, blk, 0, stream>>>(wg, wgb, EDF);
    k_cast_bf16<<<gcast, blk, 0, stream>>>(wu, wub, EDF);
    k_cast_bf16<<<gcast, blk, 0, stream>>>(wd, wdb, EDF);

    // ---- attention sub-block ----
    k_layernorm<<<SQ, blk, 0, stream>>>(hs, ln1w, ln1b, x1b);

    const dim3 gQKV(DM / BN, SQ / 256);    // N=1024, M=2048, BM=256
    k_gemm<2, false><<<gQKV, blk, 0, stream>>>(x1b, DM, wqb, DM, qf, DM, SQ, DM, DM, 1.0f, nullptr, 0, 0);
    k_gemm<2, false><<<gQKV, blk, 0, stream>>>(x1b, DM, wkb, DM, kf, DM, SQ, DM, DM, 1.0f, nullptr, 0, 0);
    k_gemm<2, false><<<gQKV, blk, 0, stream>>>(x1b, DM, wvb, DM, vf, DM, SQ, DM, DM, 1.0f, nullptr, 0, 0);

    k_rope<<<SQ, blk, 0, stream>>>(qf, kf, pos, qb, kb);
    k_cast_bf16<<<gcast, blk, 0, stream>>>(vf, vb, SD);

    const float scale = 1.0f / sqrtf((float)HD);
    const dim3 gScore(SQ / BN, SQ / 256);  // N=2048, M=2048, BM=256
    const dim3 gPV(HD / BN, SQ / 128);     // N=64,   M=2048, BM=128 (MREP=1)
    for (int h = 0; h < NH; ++h) {
        const size_t ho = (size_t)h * HD;
        // scores = scale * q_h @ k_h^T   (TRANS_B path, K=64)
        k_gemm<2, true><<<gScore, blk, 0, stream>>>(qb + ho, DM, kb + ho, DM,
                                                    scores, SQ, SQ, SQ, HD, scale, nullptr, 0, 0);
        k_softmax_causal<<<SQ, blk, 0, stream>>>(scores, probs);
        // attnO_h = P @ v_h   (K=2048, N=64)
        k_gemm<1, false><<<gPV, blk, 0, stream>>>(probs, SQ, vb + ho, DM,
                                                  attnO + ho, DM, SQ, HD, SQ, 1.0f, nullptr, 0, 0);
    }

    k_cast_bf16<<<gcast, blk, 0, stream>>>(attnO, attnOb, SD);
    k_copy_f32<<<gcast, blk, 0, stream>>>(hs, hs1, SD);
    // hs1 += attnO @ wo
    k_gemm<2, false><<<gQKV, blk, 0, stream>>>(attnOb, DM, wob, DM, hs1, DM, SQ, DM, DM, 1.0f, nullptr, 0, 1);

    // ---- MoE sub-block ----
    k_layernorm<<<SQ, blk, 0, stream>>>(hs1, ln2w, ln2b, x2b);
    k_zero_f32<<<1, 32, 0, stream>>>(red, 32);
    k_router<<<SQ, blk, 0, stream>>>(x2b, wr, combine, red);
    k_zero_f32<<<gcast, blk, 0, stream>>>(moe, SD);

    const dim3 gFF(FF / BN, SQ / 256);     // N=2048, M=2048
    const dim3 gDown(DM / BN, SQ / 256);   // N=1024, M=2048
    for (int e = 0; e < NE; ++e) {
        const bf16* wge = wgb + (size_t)e * DM * FF;
        const bf16* wue = wub + (size_t)e * DM * FF;
        const bf16* wde = wdb + (size_t)e * FF * DM;
        k_gemm<2, false><<<gFF, blk, 0, stream>>>(x2b, DM, wge, FF, gbuf, FF, SQ, FF, DM, 1.0f, nullptr, 0, 0);
        k_gemm<2, false><<<gFF, blk, 0, stream>>>(x2b, DM, wue, FF, ubuf, FF, SQ, FF, DM, 1.0f, nullptr, 0, 0);
        k_silu_mul<<<gcast, blk, 0, stream>>>(gbuf, ubuf, hb, SF);
        // moe += combine[:,e] * (h @ wd_e)   -- fused row-scale + accumulate
        k_gemm<2, false><<<gDown, blk, 0, stream>>>(hb, FF, wde, DM, moe, DM, SQ, DM, FF, 1.0f,
                                                    combine + e, NE, 1);
    }

    k_finalize<<<gcast, blk, 0, stream>>>(hs1, moe, out, SD);
    k_aux<<<1, 1, 0, stream>>>(red, out);
}
